// MultiHeadedAttention_71579924955678
// MI455X (gfx1250) — matrix operand order repaired and checked by reading, not compiled, not run on hardware
//
#include <hip/hip_runtime.h>

// ---------------------------------------------------------------------------
// MultiHeadedAttention for MI455X (gfx1250, wave32, WMMA f16 16x16x32)
// B=2, S=2048, D=1024, H=16, DK=64
//
// Pipeline:
//   1) qkv_gemm<0>: Qh = (query @ Wq^T + bq)  -> f16 [B,H,S,DK]
//      qkv_gemm<0>: Kh = (key   @ Wk^T + bk)  -> f16 [B,H,S,DK]
//      qkv_gemm<2>: Vt = (value @ Wv^T + bv)  -> f16 [B,H,DK,S] (transposed)
//   2) attn_kernel: single-pass flash attention, post-softmax group_prob
//      scaling, f16 WMMA QK^T and PV, f32 accumulation -> Ah f32 [B*S, D]
//   3) qkv_gemm<1>: d_out = Ah @ Wo^T + bo -> f32
//
// Workspace: 3 * 8MB (f16 Q/K/Vt) + 16MB (f32 Ah) = 40MB.
// ---------------------------------------------------------------------------

typedef __attribute__((ext_vector_type(16))) _Float16 v16h;
typedef __attribute__((ext_vector_type(8)))  _Float16 v8h;
typedef __attribute__((ext_vector_type(8)))  float    v8f;

union V16 { v16h v; v8h h[2]; };
union F4x2 { float4 f[2]; float a[8]; };
union I4x2 { int4 i[2]; int a[8]; };

#define WMMA_F16(A, B, C) \
  __builtin_amdgcn_wmma_f32_16x16x32_f16(false, (A), false, (B), (short)0, (C), false, false)

// lane <-> lane^16 exchange via ds_swizzle SWAPX16 (xor=0x10, and=0x1f -> imm 0x401F)
__device__ inline float xor16(float x) {
  int r = __builtin_amdgcn_ds_swizzle(__builtin_bit_cast(int, x), 0x401F);
  return __builtin_bit_cast(float, r);
}

__device__ inline v8h cvt8(float4 a, float4 b) {
  v8h h;
  h[0] = (_Float16)a.x; h[1] = (_Float16)a.y; h[2] = (_Float16)a.z; h[3] = (_Float16)a.w;
  h[4] = (_Float16)b.x; h[5] = (_Float16)b.y; h[6] = (_Float16)b.z; h[7] = (_Float16)b.w;
  return h;
}

// ---------------------------------------------------------------------------
// GEMM: C[M=4096, N=1024] = X[M,K=1024] * W[N,K]^T + bias
// Block tile 128x128, K-step 64. 8 waves: wave (wm, wn) owns 32(M) x 64(N),
// 16 WMMAs per wave per K-step, 16 K-steps.
// MODE 0: out f16 [B,H,S,DK]; MODE 2: out f16 [B,H,DK,S]; MODE 1: out f32 [M,N]
// ---------------------------------------------------------------------------
template <int MODE>
__global__ __launch_bounds__(256) void qkv_gemm(
    const float* __restrict__ X, const float* __restrict__ W,
    const float* __restrict__ bias, _Float16* __restrict__ outH,
    float* __restrict__ outF)
{
  __shared__ __align__(16) _Float16 As[128 * 72];
  __shared__ __align__(16) _Float16 Bs[128 * 72];

  const int tid  = threadIdx.x;
  const int lane = tid & 31;
  const int wid  = tid >> 5;
  const int l16  = lane & 15;
  const int hh   = lane >> 4;            // half-wave selector
  const int m0   = (wid & 3) * 32;       // wave M offset in block tile
  const int n0   = (wid >> 2) * 64;      // wave N offset in block tile
  const int mblk = blockIdx.y * 128;
  const int nblk = blockIdx.x * 128;

  const int lrow = tid >> 1;             // 0..127  cooperative-load row
  const int lcol = (tid & 1) * 32;       // 0 or 32 cooperative-load col

  const float* __restrict__ xrow = X + (size_t)(mblk + lrow) * 1024 + lcol;
  const float* __restrict__ wrow = W + (size_t)(nblk + lrow) * 1024 + lcol;

  v8f acc[2][4];
#pragma unroll
  for (int i = 0; i < 2; ++i)
#pragma unroll
    for (int j = 0; j < 4; ++j) acc[i][j] = (v8f){0.f,0.f,0.f,0.f,0.f,0.f,0.f,0.f};

  for (int k0 = 0; k0 < 1024; k0 += 64) {
    // -- prefetch next K-step tiles into cache ------------------------------
    if (k0 + 64 < 1024) {
      __builtin_prefetch(xrow + k0 + 64);
      __builtin_prefetch(wrow + k0 + 64);
    }
    // -- stage A (X) and B (W) tiles into LDS as f16 ------------------------
    {
      const float4* sa = (const float4*)(xrow + k0);
#pragma unroll
      for (int u = 0; u < 4; ++u) {
        float4 a0 = sa[2 * u], a1 = sa[2 * u + 1];
        *(v8h*)&As[lrow * 72 + lcol + 8 * u] = cvt8(a0, a1);
      }
      const float4* sb = (const float4*)(wrow + k0);
#pragma unroll
      for (int u = 0; u < 4; ++u) {
        float4 b0 = sb[2 * u], b1 = sb[2 * u + 1];
        *(v8h*)&Bs[lrow * 72 + lcol + 8 * u] = cvt8(b0, b1);
      }
    }
    __syncthreads();

    // -- fragments per documented CDNA5 layouts, 2 dk chunks of 32 ----------
#pragma unroll
    for (int c = 0; c < 2; ++c) {
      V16 af[2], bf[4];
#pragma unroll
      for (int i = 0; i < 2; ++i) {
        const _Float16* p = &As[(m0 + i * 16 + l16) * 72 + c * 32];
        af[i].h[0] = *(const v8h*)(p + hh * 8);        // K = h*8 + e
        af[i].h[1] = *(const v8h*)(p + 16 + hh * 8);   // K = 16 + h*8 + e
      }
#pragma unroll
      for (int j = 0; j < 4; ++j) {
        const _Float16* p = &Bs[(n0 + j * 16 + l16) * 72 + c * 32 + hh * 8];
        bf[j].h[0] = *(const v8h*)(p);                 // K = h*8 + e
        bf[j].h[1] = *(const v8h*)(p + 16);
      }
#pragma unroll
      for (int i = 0; i < 2; ++i)
#pragma unroll
        for (int j = 0; j < 4; ++j)
          acc[i][j] = WMMA_F16(af[i].v, bf[j].v, acc[i][j]);
    }
    __syncthreads();
  }

  // -- epilogue: C layout lane l16 = N col, VGPR r = row r + 8*hh -----------
#pragma unroll
  for (int j = 0; j < 4; ++j) {
    const int dcol = nblk + n0 + j * 16 + l16;
    const float bz = bias[dcol];
    const int head = dcol >> 6, dk = dcol & 63;
#pragma unroll
    for (int i = 0; i < 2; ++i) {
#pragma unroll
      for (int r = 0; r < 8; ++r) {
        const int mrow = mblk + m0 + i * 16 + r + hh * 8;
        const float val = acc[i][j][r] + bz;
        if (MODE == 0) {
          const int bb = mrow >> 11, ss = mrow & 2047;
          outH[(((bb * 16 + head) * 2048) + ss) * 64 + dk] = (_Float16)val;
        } else if (MODE == 2) {
          const int bb = mrow >> 11, ss = mrow & 2047;
          outH[(((bb * 16 + head) * 64) + dk) * 2048 + ss] = (_Float16)val;
        } else {
          outF[(size_t)mrow * 1024 + dcol] = val;
        }
      }
    }
  }
}

// ---------------------------------------------------------------------------
// Flash attention: grid (S/128, H, B), 256 threads = 8 waves, wave = 16 q rows.
// Per 32-key chunk: 4 WMMAs -> scores^T (rows=keys, cols=q); online softmax
// with per-wave LDS stats (no cross-wave sync in hot loop); P staged via LDS;
// 4 WMMAs -> O accumulation over dk (Vt is [B,H,DK,S] so B-frags are
// two contiguous 16B loads).
// ---------------------------------------------------------------------------
__global__ __launch_bounds__(256) void attn_kernel(
    const _Float16* __restrict__ Qh, const _Float16* __restrict__ Kh,
    const _Float16* __restrict__ Vt, const float* __restrict__ gp,
    const int* __restrict__ mask, float* __restrict__ Ah)
{
  __shared__ int msk[2048];
  __shared__ __align__(16) _Float16 Pl[8][16 * 40];
  __shared__ float stm[8][16], stz[8][16], sts[8][16];

  const int tid  = threadIdx.x;
  const int lane = tid & 31;
  const int w    = tid >> 5;
  const int l16  = lane & 15;
  const int hh   = lane >> 4;
  const int q0   = blockIdx.x * 128;
  const int hd   = blockIdx.y;
  const int b    = blockIdx.z;
  const int bh   = b * 16 + hd;
  const int qbase = q0 + w * 16;
  const int q_g   = qbase + l16;

  for (int i = tid; i < 2048; i += 256) msk[i] = mask[b * 2048 + i];
  if (lane < 16) { stm[w][l16] = -1e30f; stz[w][l16] = 0.f; sts[w][l16] = 0.f; }
  __syncthreads();

  volatile float* vm = stm[w];
  volatile float* vz = stz[w];
  volatile float* vs = sts[w];

  const _Float16* __restrict__ kbase = Kh + (size_t)bh * 2048 * 64;
  const _Float16* __restrict__ vbase = Vt + (size_t)bh * 64 * 2048;
  const float*    __restrict__ grow0 = gp + ((size_t)b * 2048 + q_g) * 2048;

  // hoisted Q B-fragments (B = Q^T: col = q, K-dim = dk, two dk chunks of 32)
  V16 qb[2];
  {
    const _Float16* qrow = Qh + ((size_t)bh * 2048 + qbase + l16) * 64;
#pragma unroll
    for (int c = 0; c < 2; ++c) {
      qb[c].h[0] = *(const v8h*)(qrow + c * 32 + hh * 8);
      qb[c].h[1] = *(const v8h*)(qrow + c * 32 + 16 + hh * 8);
    }
  }

  v8f oacc[4];
#pragma unroll
  for (int j = 0; j < 4; ++j) oacc[j] = (v8f){0.f,0.f,0.f,0.f,0.f,0.f,0.f,0.f};

  for (int kb = 0; kb < 2048; kb += 32) {
    // ---- prefetch next chunk's K rows and group_prob ---------------------
    if (kb + 32 < 2048) {
      __builtin_prefetch(kbase + (size_t)(kb + 32 + lane) * 64);
      __builtin_prefetch(grow0 + kb + 32);
    }

    // ---- scores^T tiles: C[key][q], key tiles t=0,1 ----------------------
    v8f sc[2];
#pragma unroll
    for (int t = 0; t < 2; ++t) {
      const _Float16* krow = kbase + (size_t)(kb + t * 16 + l16) * 64;
      v8f a = (v8f){0.f,0.f,0.f,0.f,0.f,0.f,0.f,0.f};
#pragma unroll
      for (int c = 0; c < 2; ++c) {
        V16 af;
        af.h[0] = *(const v8h*)(krow + c * 32 + hh * 8);
        af.h[1] = *(const v8h*)(krow + c * 32 + 16 + hh * 8);
        a = WMMA_F16(af.v, qb[c].v, a);
      }
      sc[t] = a;
    }

    // ---- mask (padding-row | diagonal) + chunk max -----------------------
    float sv[16];
    float cmax = -1e30f;
#pragma unroll
    for (int t = 0; t < 2; ++t) {
      I4x2 mk;
      mk.i[0] = *(const int4*)&msk[kb + t * 16 + hh * 8];
      mk.i[1] = *(const int4*)&msk[kb + t * 16 + hh * 8 + 4];
#pragma unroll
      for (int r = 0; r < 8; ++r) {
        const int key_g = kb + t * 16 + hh * 8 + r;
        const float s = (mk.a[r] != 0 || key_g == q_g) ? sc[t][r] * 0.125f : -1e9f;
        sv[t * 8 + r] = s;
        cmax = fmaxf(cmax, s);
      }
    }
    cmax = fmaxf(cmax, xor16(cmax));            // combine key halves (lane ^ 16)

    const float m_old = vm[l16];
    const float m_new = fmaxf(m_old, cmax);
    const float rescale = __expf(m_old - m_new);

    // ---- exp, Z partial, P = exp * group_prob (post-softmax scale) -------
    float zp = 0.f;
#pragma unroll
    for (int t = 0; t < 2; ++t) {
      const float* grow = grow0 + kb + t * 16 + hh * 8;
      F4x2 g;
      g.f[0] = *(const float4*)grow;
      g.f[1] = *(const float4*)(grow + 4);
      v8h ph;
#pragma unroll
      for (int r = 0; r < 8; ++r) {
        const float e = __expf(sv[t * 8 + r] - m_new);
        zp += e;
        ph[r] = (_Float16)(e * g.a[r]);
      }
      *(v8h*)&Pl[w][l16 * 40 + t * 16 + hh * 8] = ph;   // P[q][key]
    }
    zp += xor16(zp);

    if (lane < 16) {
      vz[l16] = vz[l16] * rescale + zp;
      vm[l16] = m_new;
      vs[l16] = rescale;
    }

    // ---- rescale O accumulators (row q = r + 8*hh) -----------------------
    float rs[8];
#pragma unroll
    for (int r = 0; r < 8; ++r) rs[r] = vs[r + hh * 8];
#pragma unroll
    for (int j = 0; j < 4; ++j)
#pragma unroll
      for (int r = 0; r < 8; ++r) oacc[j][r] *= rs[r];

    // ---- PV: A = P [16q x 32key] from LDS, B = Vt slices -----------------
    V16 pf;
    {
      const _Float16* prow = &Pl[w][l16 * 40];
      pf.h[0] = *(const v8h*)(prow + hh * 8);
      pf.h[1] = *(const v8h*)(prow + 16 + hh * 8);
    }
#pragma unroll
    for (int j = 0; j < 4; ++j) {
      const _Float16* vrow = vbase + (size_t)(j * 16 + l16) * 2048 + kb + hh * 8;
      V16 bv;
      bv.h[0] = *(const v8h*)vrow;
      bv.h[1] = *(const v8h*)(vrow + 16);
      oacc[j] = WMMA_F16(pf.v, bv.v, oacc[j]);
    }
  }

  // ---- finalize: divide by Z (Z excludes group_prob, per reference) ------
  float zi[8];
#pragma unroll
  for (int r = 0; r < 8; ++r) zi[r] = 1.f / vz[r + hh * 8];
#pragma unroll
  for (int j = 0; j < 4; ++j) {
    const int dcol = hd * 64 + j * 16 + l16;
#pragma unroll
    for (int r = 0; r < 8; ++r) {
      const int srow = qbase + r + hh * 8;
      Ah[((size_t)b * 2048 + srow) * 1024 + dcol] = oacc[j][r] * zi[r];
    }
  }
}

// ---------------------------------------------------------------------------
extern "C" void kernel_launch(void* const* d_in, const int* in_sizes, int n_in,
                              void* d_out, int out_size, void* d_ws, size_t ws_size,
                              hipStream_t stream) {
  (void)in_sizes; (void)n_in; (void)out_size; (void)ws_size;

  const float* query = (const float*)d_in[0];
  const float* key   = (const float*)d_in[1];
  const float* value = (const float*)d_in[2];
  const float* gp    = (const float*)d_in[3];
  const int*   mask  = (const int*)d_in[4];
  const float* Wq = (const float*)d_in[5];
  const float* bq = (const float*)d_in[6];
  const float* Wk = (const float*)d_in[7];
  const float* bk = (const float*)d_in[8];
  const float* Wv = (const float*)d_in[9];
  const float* bv = (const float*)d_in[10];
  const float* Wo = (const float*)d_in[11];
  const float* bo = (const float*)d_in[12];

  const size_t QKV_BYTES = (size_t)2 * 16 * 2048 * 64 * sizeof(_Float16); // 8 MB
  char* ws = (char*)d_ws;
  _Float16* Qh = (_Float16*)(ws);
  _Float16* Kh = (_Float16*)(ws + QKV_BYTES);
  _Float16* Vt = (_Float16*)(ws + 2 * QKV_BYTES);
  float*    Ah = (float*)(ws + 3 * QKV_BYTES);                           // 16 MB

  dim3 gg(8, 32), gb(256);
  qkv_gemm<0><<<gg, gb, 0, stream>>>(query, Wq, bq, Qh, nullptr);
  qkv_gemm<0><<<gg, gb, 0, stream>>>(key,   Wk, bk, Kh, nullptr);
  qkv_gemm<2><<<gg, gb, 0, stream>>>(value, Wv, bv, Vt, nullptr);

  attn_kernel<<<dim3(16, 16, 2), 256, 0, stream>>>(Qh, Kh, Vt, gp, mask, Ah);

  qkv_gemm<1><<<gg, gb, 0, stream>>>(Ah, Wo, bo, nullptr, (float*)d_out);
}
